// AdaptiveJacobianPrunedViT_72696616452805
// MI455X (gfx1250) — matrix operand order repaired and checked by reading, not compiled, not run on hardware
//
#include <hip/hip_runtime.h>
#include <hip/hip_bf16.h>
#include <stdint.h>

typedef __bf16 bf16_t;
typedef __attribute__((ext_vector_type(8)))  bf16_t bfx8;
typedef __attribute__((ext_vector_type(16))) bf16_t bfx16;
typedef __attribute__((ext_vector_type(8)))  float  fx8;

#ifndef __has_builtin
#define __has_builtin(x) 0
#endif
#if __has_builtin(__builtin_amdgcn_global_load_async_to_lds_b128) && \
    __has_builtin(__builtin_amdgcn_s_wait_asynccnt)
#define USE_ASYNC_COPY 1
#else
#define USE_ASYNC_COPY 0
#endif

#if USE_ASYNC_COPY
typedef int v4i __attribute__((__vector_size__(16)));
typedef __attribute__((address_space(1))) v4i* gv4i_p;   // global int4*
typedef __attribute__((address_space(3))) v4i* lv4i_p;   // LDS int4*
#endif

// ---------------- problem constants ----------------
constexpr int Bz   = 8;
constexpr int Cc   = 3;
constexpr int IMGs = 384;
constexpr int Pp   = 16;
constexpr int Gg   = 24;          // patches per side
constexpr int NP   = Gg * Gg;     // 576 patch tokens
constexpr int TMAX = NP + 1;      // 577 tokens incl cls
constexpr int Dd   = 384;
constexpr int Hh   = 6;
constexpr int HD   = 64;
constexpr int Ll   = 12;
constexpr int MLPd = 1536;
constexpr int NCLS = 1000;
constexpr int NCLSP= 1008;        // padded to mult of 16
constexpr int TP   = 592;         // token rows padded (mult 16)
constexpr int TK   = 640;         // key dim padded (mult 64 for BK=64)
constexpr int PKK  = Cc * Pp * Pp; // 768

// =====================================================================
// bf16 WMMA GEMM, LDS-staged, async double-buffered, BK=64.
//   C[z] = A[z] (MxK row-major) * Bt[z]^T + bias   (Bt stored N x K)
// Block tile 128(M) x 64(N); 8 waves arranged 4(M) x 2(N); each wave
// owns a 32x32 output (2x2 WMMA tiles) -> 8 WMMAs per K-step.
// K must be a multiple of 64. A has M rows, Bt has N rows allocated
// (padded, always finite) -> staging clamps rows, no compute guards;
// stores masked by Mvalid/Nvalid.
// C batch offset = (z/zInner)*sCo + (z%zInner)*sCi
// =====================================================================
__global__ __launch_bounds__(256)
void gemm_bf16_wmma(const bf16_t* __restrict__ A, long long sA, int lda,
                    const bf16_t* __restrict__ Bt, long long sB, int ldb,
                    const float* __restrict__ bias,
                    float* __restrict__ C, int zInner, long long sCo, long long sCi, int ldc,
                    int M, int N, int K, int act, int Mvalid, int Nvalid)
{
  __shared__ __align__(16) bf16_t lA[2][128 * 64];   // 16KB x2
  __shared__ __align__(16) bf16_t lB[2][64 * 64];    //  8KB x2

  const int tid  = threadIdx.x;
  const int lane = tid & 31;
  const int w    = tid >> 5;
  const int wm   = w >> 1;              // 0..3
  const int wn   = w & 1;               // 0..1
  const int z    = blockIdx.z;
  const int mBase = blockIdx.y * 128;
  const int nBase = blockIdx.x * 64;
  const int half = lane >> 4;
  const int l16  = lane & 15;

  const bf16_t* Ab = A + (long long)z * sA;
  const bf16_t* Bb = Bt + (long long)z * sB;

  auto stage = [&](int buf, int k0) {
    // A tile: 128 rows x 128B = 1024 16B-chunks -> 4 per thread
#pragma unroll
    for (int it = 0; it < 4; ++it) {
      int chunk = tid + it * 256;
      int r  = chunk >> 3;
      int co = (chunk & 7) * 8;
      int row = mBase + r; if (row >= M) row = M - 1;
      const bf16_t* g = Ab + (long long)row * lda + k0 + co;
#if USE_ASYNC_COPY
      __builtin_amdgcn_global_load_async_to_lds_b128(
          (gv4i_p)(void*)g, (lv4i_p)(void*)&lA[buf][r * 64 + co], 0, 0);
#else
      *(bfx8*)&lA[buf][r * 64 + co] = *(const bfx8*)g;
#endif
    }
    // B tile: 64 rows x 128B = 512 chunks -> 2 per thread
#pragma unroll
    for (int it = 0; it < 2; ++it) {
      int chunk = tid + it * 256;
      int r  = chunk >> 3;
      int co = (chunk & 7) * 8;
      int row = nBase + r; if (row >= N) row = N - 1;
      const bf16_t* g = Bb + (long long)row * ldb + k0 + co;
#if USE_ASYNC_COPY
      __builtin_amdgcn_global_load_async_to_lds_b128(
          (gv4i_p)(void*)g, (lv4i_p)(void*)&lB[buf][r * 64 + co], 0, 0);
#else
      *(bfx8*)&lB[buf][r * 64 + co] = *(const bfx8*)g;
#endif
    }
  };

  const int mo = wm * 32;
  const int no = wn * 32;

  fx8 acc[2][2];
  const fx8 fz = {0.f,0.f,0.f,0.f,0.f,0.f,0.f,0.f};
#pragma unroll
  for (int i = 0; i < 2; ++i)
#pragma unroll
    for (int j = 0; j < 2; ++j) acc[i][j] = fz;

  stage(0, 0);
  int buf = 0;
  for (int k0 = 0; k0 < K; k0 += 64) {
    if (k0 + 64 < K) {
      stage(buf ^ 1, k0 + 64);
#if USE_ASYNC_COPY
      __builtin_amdgcn_s_wait_asynccnt((short)6);   // prev stage drained (in-order)
#endif
    } else {
#if USE_ASYNC_COPY
      __builtin_amdgcn_s_wait_asynccnt((short)0);
#endif
    }
    __syncthreads();

#pragma unroll
    for (int kk = 0; kk < 64; kk += 32) {
      bfx16 af[2], bfr[2];
#pragma unroll
      for (int i = 0; i < 2; ++i) {
        const bf16_t* pa = &lA[buf][(mo + i * 16 + l16) * 64 + kk];
        bfx8 a0 = *(const bfx8*)(pa + half * 8);
        bfx8 a1 = *(const bfx8*)(pa + half * 8 + 16);
        const bf16_t* pb = &lB[buf][(no + i * 16 + l16) * 64 + kk];
        bfx8 b0 = *(const bfx8*)(pb + half * 8);
        bfx8 b1 = *(const bfx8*)(pb + half * 8 + 16);
#pragma unroll
        for (int e = 0; e < 8; ++e) {
          af[i][e] = a0[e];  af[i][8 + e] = a1[e];
          bfr[i][e] = b0[e]; bfr[i][8 + e] = b1[e];
        }
      }
#pragma unroll
      for (int i = 0; i < 2; ++i)
#pragma unroll
        for (int j = 0; j < 2; ++j)
          acc[i][j] = __builtin_amdgcn_wmma_f32_16x16x32_bf16(
              false, af[i], false, bfr[j], (short)0, acc[i][j], false, false);
    }

    __syncthreads();
    buf ^= 1;
  }

  // epilogue (store-masked; out-of-range sub-tiles hold finite garbage)
  float* pC = C + (long long)(z / zInner) * sCo + (long long)(z % zInner) * sCi;
#pragma unroll
  for (int j = 0; j < 2; ++j) {
    const int n = nBase + no + j * 16 + l16;
    if (n >= Nvalid) continue;
    const float bv = bias ? bias[n] : 0.f;
#pragma unroll
    for (int i = 0; i < 2; ++i) {
#pragma unroll
      for (int v = 0; v < 8; ++v) {
        int m = mBase + mo + i * 16 + v + half * 8;
        if (m < Mvalid) {
          float x = acc[i][j][v] + bv;
          if (act == 1) x = 0.5f * x * (1.f + erff(x * 0.70710678118654752f));
          pC[(long long)m * ldc + n] = x;
        }
      }
    }
  }
}

// ---------------- LayerNorm -> bf16 rows (zero beyond live tokens) -----------
__global__ void k_ln_bf16(const float* __restrict__ x, const float* __restrict__ s,
                          const float* __restrict__ b, bf16_t* __restrict__ out,
                          const int* __restrict__ nTok)
{
  __shared__ float red[128];
  const int tid = threadIdx.x;
  const int row = blockIdx.x;            // Bz*TP rows
  const int bb = row / TP, t = row % TP;
  const int Tcur = *nTok + 1;
  bf16_t* orow = out + (long long)row * Dd;
  if (t >= Tcur) { for (int d = tid; d < Dd; d += 128) orow[d] = (bf16_t)0.f; return; }
  const float* xr = x + ((long long)bb * TMAX + t) * Dd;
  float sm = 0.f, sq = 0.f;
  for (int d = tid; d < Dd; d += 128) { float v = xr[d]; sm += v; sq += v * v; }
  red[tid] = sm; __syncthreads();
  for (int st = 64; st > 0; st >>= 1) { if (tid < st) red[tid] += red[tid + st]; __syncthreads(); }
  const float mean = red[0] / Dd; __syncthreads();
  red[tid] = sq; __syncthreads();
  for (int st = 64; st > 0; st >>= 1) { if (tid < st) red[tid] += red[tid + st]; __syncthreads(); }
  const float var  = red[0] / Dd - mean * mean;
  const float rinv = rsqrtf(var + 1e-5f);
  for (int d = tid; d < Dd; d += 128)
    orow[d] = (bf16_t)((xr[d] - mean) * rinv * s[d] + b[d]);
}

// ---------------- final LN of cls rows -> (16,384) bf16 ----------------------
__global__ void k_ln_head(const float* __restrict__ x, const float* __restrict__ s,
                          const float* __restrict__ b, bf16_t* __restrict__ out)
{
  __shared__ float red[128];
  const int tid = threadIdx.x;
  const int bb = blockIdx.x;             // 16 blocks
  bf16_t* orow = out + bb * Dd;
  if (bb >= Bz) { for (int d = tid; d < Dd; d += 128) orow[d] = (bf16_t)0.f; return; }
  const float* xr = x + (long long)bb * TMAX * Dd;   // token 0
  float sm = 0.f, sq = 0.f;
  for (int d = tid; d < Dd; d += 128) { float v = xr[d]; sm += v; sq += v * v; }
  red[tid] = sm; __syncthreads();
  for (int st = 64; st > 0; st >>= 1) { if (tid < st) red[tid] += red[tid + st]; __syncthreads(); }
  const float mean = red[0] / Dd; __syncthreads();
  red[tid] = sq; __syncthreads();
  for (int st = 64; st > 0; st >>= 1) { if (tid < st) red[tid] += red[tid + st]; __syncthreads(); }
  const float var  = red[0] / Dd - mean * mean;
  const float rinv = rsqrtf(var + 1e-5f);
  for (int d = tid; d < Dd; d += 128)
    orow[d] = (bf16_t)((xr[d] - mean) * rinv * s[d] + b[d]);
}

// ---------------- split qkv -> q,k (row-major bf16), vT (d-major bf16) -------
__global__ void k_split_qkv(const float* __restrict__ qkv, bf16_t* __restrict__ q,
                            bf16_t* __restrict__ k, bf16_t* __restrict__ vT,
                            const int* __restrict__ nTok)
{
  int idx = blockIdx.x * 256 + threadIdx.x;
  if (idx >= TK * HD) return;
  int t = idx / HD, d = idx % HD;
  int b = blockIdx.z, h = blockIdx.y;
  int Tcur = *nTok + 1;
  float qv = 0.f, kv = 0.f, vv = 0.f;
  if (t < Tcur) {
    const float* base = qkv + ((long long)b * TP + t) * (3 * Dd);
    qv = base[h * HD + d];
    kv = base[Dd + h * HD + d];
    vv = base[2 * Dd + h * HD + d];
  }
  long long bh = b * Hh + h;
  q [bh * (long long)TK * HD + (long long)t * HD + d] = (bf16_t)qv;
  k [bh * (long long)TK * HD + (long long)t * HD + d] = (bf16_t)kv;
  vT[bh * (long long)HD * TK + (long long)d * TK + t] = (bf16_t)vv;
}

// ---------------- softmax over keys, bf16 probs, f32 cls row -----------------
__global__ void k_softmax(const float* __restrict__ S, bf16_t* __restrict__ P,
                          float* __restrict__ clsrow, const int* __restrict__ nTok)
{
  __shared__ float red[128];
  const int tid = threadIdx.x;
  const int row = blockIdx.x;            // BH*TP rows
  const int q = row % TP, bh = row / TP;
  const int Tcur = *nTok + 1;
  bf16_t* prow = P + (long long)row * TK;
  if (q >= Tcur) { for (int kk = tid; kk < TK; kk += 128) prow[kk] = (bf16_t)0.f; return; }
  const float* srow = S + (long long)row * TK;
  float mx = -3.0e38f;
  for (int kk = tid; kk < Tcur; kk += 128) mx = fmaxf(mx, srow[kk] * 0.125f);
  red[tid] = mx; __syncthreads();
  for (int st = 64; st > 0; st >>= 1) { if (tid < st) red[tid] = fmaxf(red[tid], red[tid + st]); __syncthreads(); }
  mx = red[0]; __syncthreads();
  float sm = 0.f;
  for (int kk = tid; kk < Tcur; kk += 128) sm += expf(srow[kk] * 0.125f - mx);
  red[tid] = sm; __syncthreads();
  for (int st = 64; st > 0; st >>= 1) { if (tid < st) red[tid] += red[tid + st]; __syncthreads(); }
  const float inv = 1.f / red[0];
  for (int kk = tid; kk < TK; kk += 128) {
    float p = (kk < Tcur) ? expf(srow[kk] * 0.125f - mx) * inv : 0.f;
    prow[kk] = (bf16_t)p;
    if (q == 0) clsrow[(long long)bh * TK + kk] = p;
  }
}

// ---------------- per (b,h,t) value-vector norms -----------------------------
__global__ void k_vnorm(const float* __restrict__ qkv, float* __restrict__ vnorm,
                        const int* __restrict__ nTok)
{
  __shared__ float red[64];
  const int tid = threadIdx.x;           // 64 threads = hd
  int id = blockIdx.x;                   // Bz*Hh*TMAX
  int t = id % TMAX; int bh = id / TMAX;
  int b = bh / Hh, h = bh % Hh;
  int Tcur = *nTok + 1;
  float v = 0.f;
  if (t < Tcur) {
    float x = qkv[((long long)b * TP + t) * (3 * Dd) + 2 * Dd + h * HD + tid];
    v = x * x;
  }
  red[tid] = v; __syncthreads();
  for (int st = 32; st > 0; st >>= 1) { if (tid < st) red[tid] += red[tid + st]; __syncthreads(); }
  if (tid == 0) vnorm[(long long)bh * TMAX + t] = sqrtf(red[0]);
}

// ---------------- entropy of cls attention row per (b,h) ---------------------
__global__ void k_entropy(const float* __restrict__ clsrow, float* __restrict__ ent,
                          const int* __restrict__ nTok)
{
  __shared__ float red[128];
  const int tid = threadIdx.x;
  const int bh = blockIdx.x;             // Bz*Hh
  const int Tcur = *nTok + 1;
  const float* p = clsrow + (long long)bh * TK;
  float s = 0.f;
  for (int kk = tid; kk < Tcur; kk += 128) s += p[kk] * logf(p[kk] + 1e-6f);
  red[tid] = s; __syncthreads();
  for (int st = 64; st > 0; st >>= 1) { if (tid < st) red[tid] += red[tid + st]; __syncthreads(); }
  if (tid == 0) ent[bh] = -red[0];
}

// ---------------- raw[b,t-1] = sum_h cls[b,h,t]*vnorm[b,h,t] -----------------
__global__ void k_raw(const float* __restrict__ clsrow, const float* __restrict__ vnorm,
                      float* __restrict__ raw, const int* __restrict__ nTok)
{
  int ti = blockIdx.x * 256 + threadIdx.x;   // [0, NP)
  if (ti >= NP) return;
  int b = blockIdx.y;
  int t = ti + 1;
  int Tcur = *nTok + 1;
  float r = 0.f;
  if (t < Tcur) {
    for (int h = 0; h < Hh; ++h) {
      long long bh = b * Hh + h;
      r += clsrow[bh * TK + t] * vnorm[bh * TMAX + t];
    }
  }
  raw[b * NP + ti] = r;
}

// ---------------- pruning decision + keep-index construction -----------------
__global__ void k_decide(const float* __restrict__ raw, const float* __restrict__ ent,
                         float* __restrict__ prevMass, int* __restrict__ prevValid,
                         int* __restrict__ nTok, int* __restrict__ keepidx)
{
  __shared__ float sSc[NP];
  __shared__ int   sKp[NP];
  __shared__ float sMass[Bz];
  __shared__ int   sNn;
  const int tid = threadIdx.x;           // blockDim = NP (576)
  const int N = *nTok;
  const int Tcur = N + 1;
  if (N <= 16) {                         // reference: skip stats entirely
    for (int i = tid; i < Tcur; i += NP) keepidx[i] = i;
    return;
  }
  if (tid < Bz) {
    float m = 0.f;
    for (int t = 0; t < NP; ++t) m += raw[tid * NP + t];
    sMass[tid] = m;
  }
  __syncthreads();
  if (tid == 0) {
    float rs = 0.f;
    for (int i = 0; i < Bz * Hh; ++i) rs += ent[i];
    float rhoMean = rs / (float)(Bz * Hh) / logf((float)Tcur);
    int Nn = N;
    if (*prevValid) {
      float ds = 0.f;
      for (int b = 0; b < Bz; ++b)
        ds += fabsf(sMass[b] - prevMass[b]) / (prevMass[b] + 1e-6f);
      float kr = 1.f - 0.1f * (rhoMean + ds / (float)Bz);
      kr = fminf(fmaxf(kr, 0.f), 1.f);
      Nn = (int)((float)N * kr);
      if (Nn < 16) Nn = 16;
    }
    for (int b = 0; b < Bz; ++b) prevMass[b] = sMass[b];
    *prevValid = 1;
    sNn = Nn;
  }
  __syncthreads();
  const int Nn = sNn;
  if (Nn >= N) {
    for (int i = tid; i < Tcur; i += NP) keepidx[i] = i;
    return;
  }
  if (tid < N) {
    float s = 0.f;
    for (int b = 0; b < Bz; ++b) s += raw[b * NP + tid] / (sMass[b] + 1e-6f);
    sSc[tid] = s / (float)Bz;
  }
  __syncthreads();
  if (tid < N) {
    float st = sSc[tid]; int rank = 0;
    for (int j = 0; j < N; ++j)
      rank += (sSc[j] > st) || (sSc[j] == st && j < tid);
    sKp[tid] = (rank < Nn) ? 1 : 0;
  }
  __syncthreads();
  if (tid < N && sKp[tid]) {
    int pos = 0;
    for (int j = 0; j < tid; ++j) pos += sKp[j];
    keepidx[1 + pos] = tid + 1;          // +1 for cls offset
  }
  if (tid == 0) { keepidx[0] = 0; *nTok = Nn; }
}

// ---------------- token gather (ping-pong x buffers) -------------------------
__global__ void k_gather(const float* __restrict__ src, float* __restrict__ dst,
                         const int* __restrict__ keepidx, const int* __restrict__ nTok)
{
  int idx = blockIdx.x * 256 + threadIdx.x;
  if (idx >= TMAX * Dd) return;
  int i = idx / Dd, d = idx % Dd;
  int b = blockIdx.y;
  int Tnew = *nTok + 1;
  if (i >= Tnew) return;
  dst[((long long)b * TMAX + i) * Dd + d] =
      src[((long long)b * TMAX + keepidx[i]) * Dd + d];
}

// ---------------- residual add (masked to live tokens) -----------------------
__global__ void k_resadd(float* __restrict__ x, const float* __restrict__ add,
                         const int* __restrict__ nTok)
{
  long long idx = (long long)blockIdx.x * 256 + threadIdx.x;
  if (idx >= (long long)Bz * TMAX * Dd) return;
  int d = (int)(idx % Dd);
  long long r = idx / Dd;
  int t = (int)(r % TMAX); int b = (int)(r / TMAX);
  if (t >= *nTok + 1) return;
  x[idx] += add[((long long)b * TP + t) * Dd + d];
}

// ---------------- f32 -> bf16 with optional row masking ----------------------
__global__ void k_f32_to_bf16(const float* __restrict__ src, bf16_t* __restrict__ dst,
                              long long total, int rowLen, int rowsPer,
                              const int* __restrict__ nTok)
{
  long long idx = (long long)blockIdx.x * 256 + threadIdx.x;
  if (idx >= total) return;
  int t = (int)((idx / rowLen) % rowsPer);
  int lim = nTok ? (*nTok + 1) : rowsPer;
  dst[idx] = (t < lim) ? (bf16_t)src[idx] : (bf16_t)0.f;
}

// ---------------- weight transpose+convert: (K,N) f32 -> (Npad,K) bf16 -------
__global__ void k_wT(const float* __restrict__ src, bf16_t* __restrict__ dst,
                     int K, int N, int Npad)
{
  long long idx = (long long)blockIdx.x * 256 + threadIdx.x;
  if (idx >= (long long)Npad * K) return;
  int n = (int)(idx / K), kk = (int)(idx % K);
  dst[idx] = (n < N) ? (bf16_t)src[(long long)kk * N + n] : (bf16_t)0.f;
}

// ---------------- extract patches -> (B*576, 768) bf16 -----------------------
__global__ void k_patches(const float* __restrict__ x, bf16_t* __restrict__ A)
{
  long long idx = (long long)blockIdx.x * 256 + threadIdx.x;
  if (idx >= (long long)Bz * NP * PKK) return;
  int e = (int)(idx % PKK);
  long long row = idx / PKK;
  int g = (int)(row % NP); int b = (int)(row / NP);
  int gy = g / Gg, gx = g % Gg;
  int c = e / (Pp * Pp); int r = (e % (Pp * Pp)) / Pp; int pc = e % Pp;
  A[idx] = (bf16_t)x[(((long long)b * Cc + c) * IMGs + gy * Pp + r) * IMGs + gx * Pp + pc];
}

// ---------------- assemble x0 = [cls; tok] + pos -----------------------------
__global__ void k_assemble(const float* __restrict__ tok, const float* __restrict__ cls,
                           const float* __restrict__ pos, float* __restrict__ x0)
{
  long long idx = (long long)blockIdx.x * 256 + threadIdx.x;
  if (idx >= (long long)Bz * TMAX * Dd) return;
  int d = (int)(idx % Dd);
  long long r = idx / Dd;
  int t = (int)(r % TMAX); int b = (int)(r / TMAX);
  float v = (t == 0) ? cls[d] : tok[((long long)b * NP + (t - 1)) * Dd + d];
  x0[idx] = v + pos[(long long)t * Dd + d];
}

__global__ void k_init(int* nTok, int* prevValid)
{
  if (threadIdx.x == 0 && blockIdx.x == 0) { *nTok = NP; *prevValid = 0; }
}

// =====================================================================
extern "C" void kernel_launch(void* const* d_in, const int* in_sizes, int n_in,
                              void* d_out, int out_size, void* d_ws, size_t ws_size,
                              hipStream_t stream)
{
  const float* in_x    = (const float*)d_in[0];
  const float* patch_w = (const float*)d_in[1];
  const float* patch_b = (const float*)d_in[2];
  const float* cls_tok = (const float*)d_in[3];
  const float* pos_emb = (const float*)d_in[4];
  const float* ln1_s   = (const float*)d_in[5];
  const float* ln1_b   = (const float*)d_in[6];
  const float* qkv_w   = (const float*)d_in[7];
  const float* qkv_b   = (const float*)d_in[8];
  const float* proj_w  = (const float*)d_in[9];
  const float* proj_b  = (const float*)d_in[10];
  const float* ln2_s   = (const float*)d_in[11];
  const float* ln2_b   = (const float*)d_in[12];
  const float* fc1_w   = (const float*)d_in[13];
  const float* fc1_b   = (const float*)d_in[14];
  const float* fc2_w   = (const float*)d_in[15];
  const float* fc2_b   = (const float*)d_in[16];
  const float* norm_s  = (const float*)d_in[17];
  const float* norm_b  = (const float*)d_in[18];
  const float* head_w  = (const float*)d_in[19];
  const float* head_b  = (const float*)d_in[20];
  float* out = (float*)d_out;

  // ---- workspace bump allocator ----
  char* base = (char*)d_ws;
  size_t off = 0;
  auto alloc = [&](size_t bytes) -> char* {
    char* p = base + off;
    off = (off + bytes + 255) & ~((size_t)255);
    return p;
  };
  bf16_t* patchT = (bf16_t*)alloc((size_t)Dd * PKK * 2);
  bf16_t* qkvT   = (bf16_t*)alloc((size_t)Ll * 3 * Dd * Dd * 2);
  bf16_t* projT  = (bf16_t*)alloc((size_t)Ll * Dd * Dd * 2);
  bf16_t* fc1T   = (bf16_t*)alloc((size_t)Ll * MLPd * Dd * 2);
  bf16_t* fc2T   = (bf16_t*)alloc((size_t)Ll * Dd * MLPd * 2);
  bf16_t* headT  = (bf16_t*)alloc((size_t)NCLSP * Dd * 2);
  float*  xb0    = (float*)alloc((size_t)Bz * TMAX * Dd * 4);
  float*  xb1    = (float*)alloc((size_t)Bz * TMAX * Dd * 4);
  bf16_t* xn     = (bf16_t*)alloc((size_t)Bz * TP * Dd * 2);
  float*  qkvO   = (float*)alloc((size_t)Bz * TP * 3 * Dd * 4);
  bf16_t* qB     = (bf16_t*)alloc((size_t)Bz * Hh * TK * HD * 2);
  bf16_t* kB     = (bf16_t*)alloc((size_t)Bz * Hh * TK * HD * 2);
  bf16_t* vT     = (bf16_t*)alloc((size_t)Bz * Hh * HD * TK * 2);
  float*  bigA   = (float*)alloc((size_t)Bz * Hh * TP * TK * 4); // scores | hidden f32 | tok
  bf16_t* bigB   = (bf16_t*)alloc((size_t)Bz * Hh * TP * TK * 2); // probs | hidden bf16 | patches
  float*  clsA   = (float*)alloc((size_t)Bz * Hh * TK * 4);
  float*  vnrm   = (float*)alloc((size_t)Bz * Hh * TMAX * 4);
  float*  aoF    = (float*)alloc((size_t)Bz * TP * Dd * 4);
  bf16_t* aoB    = (bf16_t*)alloc((size_t)Bz * TP * Dd * 2);
  float*  pout   = (float*)alloc((size_t)Bz * TP * Dd * 4);
  float*  entb   = (float*)alloc((size_t)Bz * Hh * 4);
  float*  rawb   = (float*)alloc((size_t)Bz * NP * 4);
  float*  pmass  = (float*)alloc((size_t)Bz * 4);
  int*    nTok   = (int*)alloc(16);
  int*    pValid = (int*)alloc(16);
  int*    keep   = (int*)alloc((size_t)TMAX * 4);
  bf16_t* headA  = (bf16_t*)alloc((size_t)16 * Dd * 2);

  bf16_t* patchA = bigB;   // prologue alias
  float*  tokF   = bigA;   // prologue alias
  float*  hidF   = bigA;   // MLP alias (scores dead by then)
  bf16_t* hidB   = bigB;   // MLP alias (probs dead by then)

  const dim3 blk(256, 1, 1);
  auto cdiv = [](long long a, long long b) -> unsigned { return (unsigned)((a + b - 1) / b); };
  auto gemm = [&](const bf16_t* A, long long sA, int lda,
                  const bf16_t* Bt, long long sB, int ldb,
                  const float* bias,
                  float* Cp, int zInner, long long sCo, long long sCi, int ldc,
                  int M, int N, int K, int act, int Mv, int Nv, int batch) {
    dim3 grid((unsigned)((N + 63) / 64), (unsigned)((M + 127) / 128), (unsigned)batch);
    gemm_bf16_wmma<<<grid, blk, 0, stream>>>(A, sA, lda, Bt, sB, ldb, bias,
                                             Cp, zInner, sCo, sCi, ldc,
                                             M, N, K, act, Mv, Nv);
  };

  // ---- prologue ----
  k_init<<<dim3(1), dim3(32), 0, stream>>>(nTok, pValid);
  k_wT<<<dim3(cdiv((long long)Dd * PKK, 256)), blk, 0, stream>>>(patch_w, patchT, PKK, Dd, Dd);
  for (int l = 0; l < Ll; ++l) {
    k_wT<<<dim3(cdiv((long long)3 * Dd * Dd, 256)), blk, 0, stream>>>(
        qkv_w + (long long)l * Dd * 3 * Dd, qkvT + (long long)l * 3 * Dd * Dd, Dd, 3 * Dd, 3 * Dd);
    k_wT<<<dim3(cdiv((long long)Dd * Dd, 256)), blk, 0, stream>>>(
        proj_w + (long long)l * Dd * Dd, projT + (long long)l * Dd * Dd, Dd, Dd, Dd);
    k_wT<<<dim3(cdiv((long long)MLPd * Dd, 256)), blk, 0, stream>>>(
        fc1_w + (long long)l * Dd * MLPd, fc1T + (long long)l * MLPd * Dd, Dd, MLPd, MLPd);
    k_wT<<<dim3(cdiv((long long)Dd * MLPd, 256)), blk, 0, stream>>>(
        fc2_w + (long long)l * MLPd * Dd, fc2T + (long long)l * Dd * MLPd, MLPd, Dd, Dd);
  }
  k_wT<<<dim3(cdiv((long long)NCLSP * Dd, 256)), blk, 0, stream>>>(head_w, headT, Dd, NCLS, NCLSP);

  k_patches<<<dim3(cdiv((long long)Bz * NP * PKK, 256)), blk, 0, stream>>>(in_x, patchA);
  gemm(patchA, 0, PKK, patchT, 0, PKK, patch_b,
       tokF, 1, 0, 0, Dd, Bz * NP, Dd, PKK, 0, Bz * NP, Dd, 1);
  k_assemble<<<dim3(cdiv((long long)Bz * TMAX * Dd, 256)), blk, 0, stream>>>(tokF, cls_tok, pos_emb, xb0);

  // ---- transformer layers ----
  float* xbuf[2] = { xb0, xb1 };
  for (int l = 0; l < Ll; ++l) {
    float* xin  = xbuf[l & 1];
    float* xout = xbuf[(l + 1) & 1];

    k_ln_bf16<<<dim3(Bz * TP), dim3(128), 0, stream>>>(xin, ln1_s + l * Dd, ln1_b + l * Dd, xn, nTok);
    gemm(xn, (long long)TP * Dd, Dd, qkvT + (long long)l * 3 * Dd * Dd, 0, Dd,
         qkv_b + l * 3 * Dd, qkvO, 1, (long long)TP * 3 * Dd, 0, 3 * Dd,
         TP, 3 * Dd, Dd, 0, TP, 3 * Dd, Bz);
    k_split_qkv<<<dim3(cdiv((long long)TK * HD, 256), Hh, Bz), blk, 0, stream>>>(qkvO, qB, kB, vT, nTok);
    k_vnorm<<<dim3(Bz * Hh * TMAX), dim3(64), 0, stream>>>(qkvO, vnrm, nTok);

    // scores = q @ k^T   (scale applied in softmax)
    gemm(qB, (long long)TK * HD, HD, kB, (long long)TK * HD, HD, nullptr,
         bigA, 1, (long long)TP * TK, 0, TK, TP, TK, HD, 0, TP, TK, Bz * Hh);
    k_softmax<<<dim3(Bz * Hh * TP), dim3(128), 0, stream>>>(bigA, bigB, clsA, nTok);
    k_entropy<<<dim3(Bz * Hh), dim3(128), 0, stream>>>(clsA, entb, nTok);
    k_raw<<<dim3(cdiv(NP, 256), Bz), blk, 0, stream>>>(clsA, vnrm, rawb, nTok);

    // out = P @ v  (per-head columns packed into (B,TP,D))
    gemm(bigB, (long long)TP * TK, TK, vT, (long long)HD * TK, TK, nullptr,
         aoF, Hh, (long long)TP * Dd, HD, Dd, TP, HD, TK, 0, TP, HD, Bz * Hh);
    k_f32_to_bf16<<<dim3(cdiv((long long)Bz * TP * Dd, 256)), blk, 0, stream>>>(
        aoF, aoB, (long long)Bz * TP * Dd, Dd, TP, nTok);
    gemm(aoB, (long long)TP * Dd, Dd, projT + (long long)l * Dd * Dd, 0, Dd,
         proj_b + l * Dd, pout, 1, (long long)TP * Dd, 0, Dd, TP, Dd, Dd, 0, TP, Dd, Bz);
    k_resadd<<<dim3(cdiv((long long)Bz * TMAX * Dd, 256)), blk, 0, stream>>>(xin, pout, nTok);

    // MLP
    k_ln_bf16<<<dim3(Bz * TP), dim3(128), 0, stream>>>(xin, ln2_s + l * Dd, ln2_b + l * Dd, xn, nTok);
    gemm(xn, (long long)TP * Dd, Dd, fc1T + (long long)l * MLPd * Dd, 0, Dd,
         fc1_b + l * MLPd, hidF, 1, (long long)TP * MLPd, 0, MLPd,
         TP, MLPd, Dd, 1, TP, MLPd, Bz);
    k_f32_to_bf16<<<dim3(cdiv((long long)Bz * TP * MLPd, 256)), blk, 0, stream>>>(
        hidF, hidB, (long long)Bz * TP * MLPd, MLPd, TP, nullptr);
    gemm(hidB, (long long)TP * MLPd, MLPd, fc2T + (long long)l * Dd * MLPd, 0, MLPd,
         fc2_b + l * Dd, pout, 1, (long long)TP * Dd, 0, Dd, TP, Dd, MLPd, 0, TP, Dd, Bz);
    k_resadd<<<dim3(cdiv((long long)Bz * TMAX * Dd, 256)), blk, 0, stream>>>(xin, pout, nTok);

    // prune + compact (identity map when no prune)
    k_decide<<<dim3(1), dim3(NP), 0, stream>>>(rawb, entb, pmass, pValid, nTok, keep);
    k_gather<<<dim3(cdiv((long long)TMAX * Dd, 256), Bz), blk, 0, stream>>>(xin, xout, keep, nTok);
  }

  // ---- head ----
  k_ln_head<<<dim3(16), dim3(128), 0, stream>>>(xbuf[0], norm_s, norm_b, headA);
  gemm(headA, 0, Dd, headT, 0, Dd, head_b,
       out, 1, 0, 0, NCLS, 16, NCLSP, Dd, 0, Bz, NCLS, 1);
}
